// BidirectionalMamba2Block_66984309948486
// MI455X (gfx1250) — compile-verified, matrix operand order checked by reading, not run on hardware
//
#include <hip/hip_runtime.h>

// ---------------------------------------------------------------------------
// Bidirectional Mamba2 block for MI455X (gfx1250, wave32, WMMA).
// GEMMs use v_wmma_f32_16x16x32_bf16 (bf16 in, f32 accum) with the weight
// tile staged into LDS through the CDNA5 async-DMA path
// (global_load_async_to_lds_b128 + s_wait_asynccnt), double-buffered so the
// DMA for tile k+1 overlaps the WMMAs on tile k. Each wave owns a 32x32
// register tile (2x2 fragments -> 4 WMMAs per K-step per barrier).
// Selective scan is register-resident per (batch, head): 64 threads, each
// holding one 64-wide state row in VGPRs, B/C broadcast via LDS per step.
// ---------------------------------------------------------------------------

constexpr int D_MODEL   = 1024;
constexpr int D_STATE   = 64;
constexpr int HEADDIM   = 64;
constexpr int D_INNER   = 2048;
constexpr int NHEADS    = 32;
constexpr int CONV_DIM  = D_INNER + 2 * D_STATE;              // 2176
constexpr int D_IN_PROJ = 2 * D_INNER + 2 * D_STATE + NHEADS; // 4256
constexpr int BATCH     = 2;
constexpr int SEQ       = 2048;
constexpr int MROWS     = BATCH * SEQ;                        // 4096
constexpr float EPSF    = 1e-5f;

typedef __bf16 bf16_t;
typedef __attribute__((ext_vector_type(16))) __bf16 v16bf;
typedef __attribute__((ext_vector_type(8)))  float  v8f;

__device__ __forceinline__ bf16_t f2bf(float f) {
  unsigned u = __builtin_bit_cast(unsigned, f);
  unsigned r = (u + 0x7FFFu + ((u >> 16) & 1u)) >> 16;   // round-to-nearest-even
  unsigned short us = (unsigned short)r;
  return __builtin_bit_cast(bf16_t, us);
}

union FragU { uint4 q[2]; v16bf v; };

// Load one 16x32 bf16 fragment half-row for this lane.
// base = row pointer advanced to k0; h = lane>>4 selects K sub-range.
// Per ISA layout: lanes 0-15 hold K={0..7,16..23}, lanes 16-31 K={8..15,24..31}.
__device__ __forceinline__ v16bf load_frag(const bf16_t* base, int h) {
  FragU f;
  f.q[0] = *reinterpret_cast<const uint4*>(base + 8 * h);        // K = 8h..8h+7
  f.q[1] = *reinterpret_cast<const uint4*>(base + 16 + 8 * h);   // K = 16+8h..
  return f.v;
}

// Raw LDS byte offset of a __shared__ object (addrspace(3) is 32-bit).
__device__ __forceinline__ unsigned lds_addr_of(const void* p) {
  return (unsigned)(unsigned long long)
         (__attribute__((address_space(3))) const char*)p;
}

// CDNA5 async DMA: 16B per lane, global -> LDS, tracked on ASYNCcnt.
__device__ __forceinline__ void async_ld_b128(unsigned lds_off, const void* gsrc) {
  asm volatile("global_load_async_to_lds_b128 %0, %1, off"
               :: "v"(lds_off), "v"(gsrc)
               : "memory");
}

__device__ __forceinline__ void wait_async0() {
  asm volatile("s_wait_asynccnt 0" ::: "memory");
}

__device__ __forceinline__ v8f wmma_bf16(v16bf a, v16bf b, v8f c) {
  return __builtin_amdgcn_wmma_f32_16x16x32_bf16(false, a, false, b,
                                                 (short)0, c, false, false);
}

// ---------------------------------------------------------------------------
// C[M,N] = A[M,K] @ W[N,K]^T   (row-major A and W; both bf16, f32 accum)
// Epilogue: optional bias[n], optional residual[m*N+n], optional sigmoid,
// optional beta-accumulate into C.
// Block = 128 threads (4 waves), block tile = 128M x 32N; wave w owns rows
// tileM + 32w as a 32x32 register tile (2x2 fragments, 4 WMMA per K-step).
// The shared 32n x 32k weight tile is staged in LDS via async-DMA (one b128
// chunk per thread per step), double-buffered.
// ---------------------------------------------------------------------------
__global__ __launch_bounds__(128)
void wmma_gemm_bf16(const bf16_t* __restrict__ A, const bf16_t* __restrict__ W,
                    float* __restrict__ C, int M, int N, int K,
                    const float* __restrict__ bias,
                    const float* __restrict__ residual,
                    int do_sigmoid, int accumulate) {
  __shared__ __align__(16) bf16_t btile[2][32][32];   // [buf][n][k] = 2 x 2KB

  const int tid   = threadIdx.x;
  const int lane  = tid & 31;
  const int wave  = tid >> 5;
  const int tileN = blockIdx.x * 32;
  const int tileM = blockIdx.y * 128 + wave * 32;
  const int nloc  = lane & 15;
  const int h     = lane >> 4;

  // async-copy assignment: each of 128 threads moves one 16B chunk per tile
  const int cp_row = tid >> 2;                    // 0..31 : n within tile
  const int cp_col = tid & 3;                     // 0..3  : 16B chunk along k
  const unsigned lds_dst0 =
      lds_addr_of(&btile[0][0][0]) + (unsigned)(cp_row * 64 + cp_col * 16);
  const bf16_t* wsrc = W + (size_t)(tileN + cp_row) * K + cp_col * 8;

  const bf16_t* Arow0 = A + (size_t)(tileM + nloc) * K;
  const bf16_t* Arow1 = A + (size_t)(tileM + 16 + nloc) * K;

  v8f acc00 = {}; v8f acc01 = {};
  v8f acc10 = {}; v8f acc11 = {};

  // prologue: stage the k0 = 0 weight tile into buffer 0
  async_ld_b128(lds_dst0, wsrc);

  int cur = 0;
  for (int k0 = 0; k0 < K; k0 += 32) {
    wait_async0();            // this wave's DMA for tile[cur] has landed
    __syncthreads();          // ...and everyone else's too
    if (k0 + 32 < K)          // overlap: DMA next tile into the other buffer
      async_ld_b128(lds_dst0 + (unsigned)((cur ^ 1) * 2048), wsrc + k0 + 32);
    if (k0 + 64 < K) {        // keep the A streams warm in cache
      __builtin_prefetch(Arow0 + k0 + 64, 0, 0);
      __builtin_prefetch(Arow1 + k0 + 64, 0, 0);
    }

    v16bf a0 = load_frag(Arow0 + k0, h);
    v16bf a1 = load_frag(Arow1 + k0, h);
    v16bf b0 = load_frag(&btile[cur][nloc][0],      h);
    v16bf b1 = load_frag(&btile[cur][16 + nloc][0], h);
    acc00 = wmma_bf16(a0, b0, acc00);
    acc01 = wmma_bf16(a0, b1, acc01);
    acc10 = wmma_bf16(a1, b0, acc10);
    acc11 = wmma_bf16(a1, b1, acc11);
    cur ^= 1;
  }

  // C/D layout: lane l holds column n = l&15; VGPR r holds row r + 8*(l>>4).
#pragma unroll
  for (int i = 0; i < 2; ++i) {
#pragma unroll
    for (int j = 0; j < 2; ++j) {
      v8f acc = (i == 0) ? (j == 0 ? acc00 : acc01)
                         : (j == 0 ? acc10 : acc11);
      int n = tileN + j * 16 + nloc;
      float bv = bias ? bias[n] : 0.0f;
#pragma unroll
      for (int r = 0; r < 8; ++r) {
        int m = tileM + i * 16 + r + 8 * h;
        size_t idx = (size_t)m * N + n;
        float v = acc[r] + bv;
        if (residual)   v += residual[idx];
        if (do_sigmoid) v = 1.0f / (1.0f + __expf(-v));
        if (accumulate) v += C[idx];
        C[idx] = v;
      }
    }
  }
}

// ---------------------------------------------------------------------------
// LayerNorm over D_MODEL, emits fp32 and bf16 copies of xn.
// ---------------------------------------------------------------------------
__global__ __launch_bounds__(256)
void layernorm_kernel(const float* __restrict__ x, const float* __restrict__ w,
                      const float* __restrict__ b, float* __restrict__ xn,
                      bf16_t* __restrict__ xn_bf) {
  const int row = blockIdx.x;
  const float* xr = x + (size_t)row * D_MODEL;
  __shared__ float red[256];

  float s = 0.f;
  for (int i = threadIdx.x; i < D_MODEL; i += 256) s += xr[i];
  red[threadIdx.x] = s; __syncthreads();
  for (int o = 128; o > 0; o >>= 1) {
    if (threadIdx.x < o) red[threadIdx.x] += red[threadIdx.x + o];
    __syncthreads();
  }
  float mu = red[0] / D_MODEL; __syncthreads();

  float v = 0.f;
  for (int i = threadIdx.x; i < D_MODEL; i += 256) {
    float d = xr[i] - mu; v += d * d;
  }
  red[threadIdx.x] = v; __syncthreads();
  for (int o = 128; o > 0; o >>= 1) {
    if (threadIdx.x < o) red[threadIdx.x] += red[threadIdx.x + o];
    __syncthreads();
  }
  float rstd = rsqrtf(red[0] / D_MODEL + EPSF);

  for (int i = threadIdx.x; i < D_MODEL; i += 256) {
    float val = (xr[i] - mu) * rstd * w[i] + b[i];
    size_t idx = (size_t)row * D_MODEL + i;
    xn[idx] = val;
    xn_bf[idx] = f2bf(val);
  }
}

// float -> bf16 weight conversion
__global__ void f2bf_kernel(const float* __restrict__ src,
                            bf16_t* __restrict__ dst, size_t n) {
  size_t i = (size_t)blockIdx.x * blockDim.x + threadIdx.x;
  if (i < n) dst[i] = f2bf(src[i]);
}

// ---------------------------------------------------------------------------
// Causal depthwise conv (width 4) + SiLU. Reads xBC slice of zxbcdt through
// the direction mapping (reverse => original index L-1-u); writes scan-order.
// ---------------------------------------------------------------------------
__global__ void conv_silu_kernel(const float* __restrict__ zxbcdt,
                                 const float* __restrict__ conv_w,
                                 const float* __restrict__ conv_b,
                                 float* __restrict__ out, int reverse) {
  size_t idx = (size_t)blockIdx.x * blockDim.x + threadIdx.x;
  size_t total = (size_t)BATCH * SEQ * CONV_DIM;
  if (idx >= total) return;
  int c = idx % CONV_DIM;
  int s = (idx / CONV_DIM) % SEQ;
  int bb = idx / ((size_t)CONV_DIM * SEQ);

  float acc = conv_b[c];
#pragma unroll
  for (int k = 0; k < 4; ++k) {
    int u = s + k - 3;
    if (u < 0) continue;
    int uo = reverse ? (SEQ - 1 - u) : u;
    acc += conv_w[c * 4 + k] *
           zxbcdt[((size_t)(bb * SEQ + uo)) * D_IN_PROJ + D_INNER + c];
  }
  float sv = acc / (1.0f + __expf(-acc));  // SiLU
  out[((size_t)(bb * SEQ + s)) * CONV_DIM + c] = sv;
}

// ---------------------------------------------------------------------------
// Selective scan. One block per (batch, head); 64 threads, thread p keeps
// h[p][0..63] in VGPRs. B/C per step staged through LDS. Output written at
// the ORIGINAL token position (un-reverses bwd direction).
// ---------------------------------------------------------------------------
__global__ __launch_bounds__(64)
void scan_kernel(const float* __restrict__ zxbcdt,
                 const float* __restrict__ xconv,
                 const float* __restrict__ dt_bias,
                 const float* __restrict__ A_log,
                 const float* __restrict__ Dp,
                 float* __restrict__ y_out, int reverse) {
  const int head = blockIdx.x % NHEADS;
  const int bb   = blockIdx.x / NHEADS;
  const int p    = threadIdx.x;

  float h[D_STATE];
#pragma unroll
  for (int n = 0; n < D_STATE; ++n) h[n] = 0.f;

  __shared__ float Bs[D_STATE];
  __shared__ float Cs[D_STATE];

  const float A   = -__expf(A_log[head]);
  const float dtb = dt_bias[head];
  const float Dh  = Dp[head];

  for (int s = 0; s < SEQ; ++s) {
    const int so = reverse ? (SEQ - 1 - s) : s;          // original token index
    const size_t rowc = ((size_t)(bb * SEQ + s)) * CONV_DIM;

    Bs[p] = xconv[rowc + D_INNER + p];
    Cs[p] = xconv[rowc + D_INNER + D_STATE + p];
    __syncthreads();

    float dtraw = zxbcdt[((size_t)(bb * SEQ + so)) * D_IN_PROJ +
                         (D_INNER + CONV_DIM) + head] + dtb;
    float dt = (dtraw > 20.f) ? dtraw : __logf(1.0f + __expf(dtraw)); // softplus
    float dA = __expf(dt * A);
    float xv = xconv[rowc + head * HEADDIM + p];
    float dtx = dt * xv;

    float y = 0.f;
#pragma unroll
    for (int n = 0; n < D_STATE; ++n) {
      h[n] = h[n] * dA + dtx * Bs[n];
      y += h[n] * Cs[n];
    }
    y_out[((size_t)(bb * SEQ + so)) * D_INNER + head * HEADDIM + p] = y + Dh * xv;
    __syncthreads();
  }
}

// ---------------------------------------------------------------------------
// Gated RMSNorm: g = y*silu(z); g *= rsqrt(mean(g^2)+eps)*norm_w; emit bf16.
// z lives in zxbcdt[:, 0:D_INNER] (natural token order).
// ---------------------------------------------------------------------------
__global__ __launch_bounds__(256)
void grms_kernel(const float* __restrict__ y, const float* __restrict__ zxbcdt,
                 const float* __restrict__ norm_w, bf16_t* __restrict__ outb) {
  const int row = blockIdx.x;
  const float* yr = y + (size_t)row * D_INNER;
  const float* zr = zxbcdt + (size_t)row * D_IN_PROJ;
  __shared__ float red[256];

  float ss = 0.f;
  for (int i = threadIdx.x; i < D_INNER; i += 256) {
    float z = zr[i];
    float g = yr[i] * (z / (1.0f + __expf(-z)));
    ss += g * g;
  }
  red[threadIdx.x] = ss; __syncthreads();
  for (int o = 128; o > 0; o >>= 1) {
    if (threadIdx.x < o) red[threadIdx.x] += red[threadIdx.x + o];
    __syncthreads();
  }
  float rms = rsqrtf(red[0] / D_INNER + EPSF);

  for (int i = threadIdx.x; i < D_INNER; i += 256) {
    float z = zr[i];
    float g = yr[i] * (z / (1.0f + __expf(-z)));
    outb[(size_t)row * D_INNER + i] = f2bf(g * rms * norm_w[i]);
  }
}

// yg = bf16(ysum * gate)
__global__ void mul_kernel(const float* __restrict__ a, const float* __restrict__ g,
                           bf16_t* __restrict__ out, size_t n) {
  size_t i = (size_t)blockIdx.x * blockDim.x + threadIdx.x;
  if (i < n) out[i] = f2bf(a[i] * g[i]);
}

// ---------------------------------------------------------------------------
// Host-side orchestration
// ---------------------------------------------------------------------------
extern "C" void kernel_launch(void* const* d_in, const int* in_sizes, int n_in,
                              void* d_out, int out_size, void* d_ws, size_t ws_size,
                              hipStream_t stream) {
  const float* x      = (const float*)d_in[0];
  const float* ln_w   = (const float*)d_in[1];
  const float* ln_b   = (const float*)d_in[2];
  const float* gate_W = (const float*)d_in[19];
  const float* gate_b = (const float*)d_in[20];
  const float* out_W  = (const float*)d_in[21];
  const float* out_b  = (const float*)d_in[22];
  float* out = (float*)d_out;

  char* ws = (char*)d_ws;
  size_t off = 0;
  auto alloc = [&](size_t bytes) -> void* {
    void* p = ws + off;
    off = (off + bytes + 255) & ~(size_t)255;
    return p;
  };

  float*  xn     = (float*) alloc((size_t)MROWS * D_MODEL * 4);
  bf16_t* xn_bf  = (bf16_t*)alloc((size_t)MROWS * D_MODEL * 2);
  bf16_t* wbuf   = (bf16_t*)alloc((size_t)D_IN_PROJ * D_MODEL * 2); // reused
  float*  zxbcdt = (float*) alloc((size_t)MROWS * D_IN_PROJ * 4);
  float*  xconv  = (float*) alloc((size_t)MROWS * CONV_DIM * 4);
  float*  ym     = (float*) alloc((size_t)MROWS * D_INNER * 4);
  bf16_t* yn_bf  = (bf16_t*)alloc((size_t)MROWS * D_INNER * 2);
  float*  ysum   = (float*) alloc((size_t)MROWS * D_MODEL * 4);
  float*  gate   = (float*) alloc((size_t)MROWS * D_MODEL * 4);
  bf16_t* yg_bf  = (bf16_t*)alloc((size_t)MROWS * D_MODEL * 2);

  layernorm_kernel<<<MROWS, 256, 0, stream>>>(x, ln_w, ln_b, xn, xn_bf);

  for (int dir = 0; dir < 2; ++dir) {
    const int base = dir ? 11 : 3;
    const float* W_in    = (const float*)d_in[base + 0];
    const float* conv_w  = (const float*)d_in[base + 1];
    const float* conv_b  = (const float*)d_in[base + 2];
    const float* dt_bias = (const float*)d_in[base + 3];
    const float* A_log   = (const float*)d_in[base + 4];
    const float* Dp      = (const float*)d_in[base + 5];
    const float* norm_w  = (const float*)d_in[base + 6];
    const float* W_out   = (const float*)d_in[base + 7];

    { size_t n = (size_t)D_IN_PROJ * D_MODEL;
      f2bf_kernel<<<(n + 255) / 256, 256, 0, stream>>>(W_in, wbuf, n); }

    { dim3 g(D_IN_PROJ / 32, MROWS / 128);   // 133 x 32 tiles
      wmma_gemm_bf16<<<g, 128, 0, stream>>>(xn_bf, wbuf, zxbcdt,
                                            MROWS, D_IN_PROJ, D_MODEL,
                                            nullptr, nullptr, 0, 0); }

    { size_t n = (size_t)BATCH * SEQ * CONV_DIM;
      conv_silu_kernel<<<(n + 255) / 256, 256, 0, stream>>>(zxbcdt, conv_w,
                                                            conv_b, xconv, dir); }

    scan_kernel<<<BATCH * NHEADS, 64, 0, stream>>>(zxbcdt, xconv, dt_bias,
                                                   A_log, Dp, ym, dir);

    grms_kernel<<<MROWS, 256, 0, stream>>>(ym, zxbcdt, norm_w, yn_bf);

    { size_t n = (size_t)D_MODEL * D_INNER;
      f2bf_kernel<<<(n + 255) / 256, 256, 0, stream>>>(W_out, wbuf, n); }

    { dim3 g(D_MODEL / 32, MROWS / 128);
      wmma_gemm_bf16<<<g, 128, 0, stream>>>(yn_bf, wbuf, ysum,
                                            MROWS, D_MODEL, D_INNER,
                                            nullptr, nullptr, 0, /*beta=*/dir); }
  }

  { size_t n = (size_t)D_MODEL * D_MODEL;
    f2bf_kernel<<<(n + 255) / 256, 256, 0, stream>>>(gate_W, wbuf, n); }
  { dim3 g(D_MODEL / 32, MROWS / 128);
    wmma_gemm_bf16<<<g, 128, 0, stream>>>(xn_bf, wbuf, gate,
                                          MROWS, D_MODEL, D_MODEL,
                                          gate_b, nullptr, /*sigmoid=*/1, 0); }

  { size_t n = (size_t)MROWS * D_MODEL;
    mul_kernel<<<(n + 255) / 256, 256, 0, stream>>>(ysum, gate, yg_bf, n); }

  { size_t n = (size_t)D_MODEL * D_MODEL;
    f2bf_kernel<<<(n + 255) / 256, 256, 0, stream>>>(out_W, wbuf, n); }
  { dim3 g(D_MODEL / 32, MROWS / 128);
    wmma_gemm_bf16<<<g, 128, 0, stream>>>(yg_bf, wbuf, out,
                                          MROWS, D_MODEL, D_MODEL,
                                          out_b, /*residual=*/x, 0, 0); }

  (void)in_sizes; (void)n_in; (void)out_size; (void)ws_size;
}
